// get_model_63642825392493
// MI455X (gfx1250) — matrix operand order repaired and checked by reading, not compiled, not run on hardware
//
#include <hip/hip_runtime.h>
#include <hip/hip_bf16.h>
#include <stdint.h>

// ---------------------------------------------------------------------------
// Types / constants
// ---------------------------------------------------------------------------
typedef _Float16 f16;
typedef __attribute__((ext_vector_type(16))) _Float16 v16h;
typedef __attribute__((ext_vector_type(8)))  _Float16 v8h;
typedef __attribute__((ext_vector_type(8)))  float    v8f;
typedef __attribute__((ext_vector_type(4)))  int      i32x4;
typedef __attribute__((ext_vector_type(8)))  int      i32x8;
typedef __attribute__((ext_vector_type(4)))  unsigned int u32x4;

#define LRELU_SLOPE 0.3f
#define BN_SCALE_F  0.9999950000374997f   // 1/sqrt(1+1e-5)

// TDM availability (device pass only; host pass takes fallback branch)
#if defined(__has_builtin)
#  if __has_builtin(__builtin_amdgcn_tensor_load_to_lds)
#    define USE_TDM 1
#  else
#    define USE_TDM 0
#  endif
#else
#  define USE_TDM 0
#endif

// Make the chosen path visible in compile stderr (device pass only).
#if defined(__HIP_DEVICE_COMPILE__) && !USE_TDM
#warning "CDNA5 TDM builtin unavailable on this toolchain: sa_mlp_kernel uses manual LDS gather fallback"
#endif

#if USE_TDM
__device__ __forceinline__ void tdm_load(u32x4 g0, i32x8 g1, i32x4 g2, i32x4 g3) {
#if defined(__clang_major__) && __clang_major__ >= 23
  i32x8 z8 = {0,0,0,0,0,0,0,0};
  __builtin_amdgcn_tensor_load_to_lds(g0, g1, g2, g3, z8, 0);
#else
  __builtin_amdgcn_tensor_load_to_lds(g0, g1, g2, g3, 0);
#endif
}
#endif

// ---------------------------------------------------------------------------
// WMMA helpers (CDNA5 v_wmma_f32_16x16x32_f16, wave32)
// A (16x32 f16, k order chosen by this code): lane<16 -> row=lane, elems 0..15 = K+0..15
//                lane>=16 -> row=lane-16, elems 0..15 = K+16..31
// B (32x16 f16, same k order as A): lane holds col = lane&15; elems i -> K = i + (lane&16 ? 16 : 0)
// C/D (16x16 f32): elem j -> M = j + (lane&16 ? 8 : 0), N = lane&15
// ---------------------------------------------------------------------------
__device__ __forceinline__ v8f wmma16(v16h a, v16h b, v8f c) {
  return __builtin_amdgcn_wmma_f32_16x16x32_f16(false, a, false, b, (short)0, c,
                                                false, false);
}

__device__ __forceinline__ v16h load_a_frag(const f16* __restrict__ W, int kstride,
                                            int mBase, int kBase, int lane) {
  int row = mBase + (lane & 15);
  int k0  = kBase + ((lane & 16) ? 16 : 0);
  const f16* p = W + (size_t)row * kstride + k0;
  v8h lo = *(const v8h*)p;          // K = k0 .. k0+7
  v8h hi = *(const v8h*)(p + 8);    // K = k0+8 .. k0+15
  return __builtin_shufflevector(lo, hi, 0,1,2,3,4,5,6,7,8,9,10,11,12,13,14,15);
}

__device__ __forceinline__ v16h load_b_frag(const f16* __restrict__ X, int kstride,
                                            int colBase, int kBase, int lane) {
  int col = colBase + (lane & 15);
  int k0  = kBase + ((lane & 16) ? 16 : 0);
  return *(const v16h*)(X + (size_t)col * kstride + k0);
}

// ---------------------------------------------------------------------------
// Generic fused GEMM: Y[col, m] = act( W[M x K] * X[col, :K] )
// X, Y are point-major fp16; act 0 = none, 1 = bn(g,b)+leaky-relu.
// One 16x16 output tile per wave, K tiled by 32.
// ---------------------------------------------------------------------------
__global__ __launch_bounds__(256)
void wmma_gemm_kernel(const f16* __restrict__ W, const f16* __restrict__ X,
                      f16* __restrict__ Y, const float* __restrict__ g,
                      const float* __restrict__ bb,
                      int M, int K, int Ncols, int act) {
  const int lane = threadIdx.x & 31;
  const int wave = threadIdx.x >> 5;
  const int nWaves = blockDim.x >> 5;
  const int nM = M >> 4, nC = Ncols >> 4, nK = K >> 5;
  const int total = nM * nC;
  for (int t = blockIdx.x * nWaves + wave; t < total; t += gridDim.x * nWaves) {
    const int mT = t % nM, cT = t / nM;
    const int mBase = mT << 4, colBase = cT << 4;
    v8f acc = {};
    const f16* wrow = W + (size_t)(mBase + (lane & 15)) * K + ((lane & 16) ? 16 : 0);
    const f16* xrow = X + (size_t)(colBase + (lane & 15)) * K + ((lane & 16) ? 16 : 0);
    for (int kt = 0; kt < nK; ++kt) {
      const int kb = kt << 5;
      __builtin_prefetch(wrow + kb + 64, 0, 3);   // near-cache prefetch of next A chunk
      v8h lo = *(const v8h*)(wrow + kb);
      v8h hi = *(const v8h*)(wrow + kb + 8);
      v16h a = __builtin_shufflevector(lo, hi, 0,1,2,3,4,5,6,7,8,9,10,11,12,13,14,15);
      v16h b = *(const v16h*)(xrow + kb);
      acc = wmma16(a, b, acc);
    }
    const int mOff = mBase + ((lane & 16) ? 8 : 0);
    const int col  = colBase + (lane & 15);
    v8h out;
    if (act == 1) {
      #pragma unroll
      for (int j = 0; j < 8; ++j) {
        float v = acc[j] * (g[mOff + j] * BN_SCALE_F) + bb[mOff + j];
        out[j] = (f16)(v >= 0.f ? v : LRELU_SLOPE * v);
      }
    } else {
      #pragma unroll
      for (int j = 0; j < 8; ++j) out[j] = (f16)acc[j];
    }
    *(v8h*)(Y + (size_t)col * M + mOff) = out;
  }
}

// ---------------------------------------------------------------------------
// Weight conversion f32 -> f16 with M/K zero padding
// ---------------------------------------------------------------------------
__global__ void cvt_pad_kernel(const float* __restrict__ src, f16* __restrict__ dst,
                               int M, int K, int Mp, int Kp) {
  int i = blockIdx.x * blockDim.x + threadIdx.x;
  if (i >= Mp * Kp) return;
  int r = i / Kp, c = i - r * Kp;
  dst[i] = (r < M && c < K) ? (f16)src[(size_t)r * K + c] : (f16)0.f;
}

// xyz (B,6,N) f32 -> xp/xn (B*N, 32) f16 zero-padded, xyzt (B*N,3) f32
__global__ void prep_inputs_kernel(const float* __restrict__ xyz, f16* __restrict__ xp,
                                   f16* __restrict__ xn, float* __restrict__ xyzt,
                                   int B, int N) {
  int i = blockIdx.x * blockDim.x + threadIdx.x;
  if (i >= B * N) return;
  int b = i / N, n = i - b * N;
  const float* base = xyz + (size_t)b * 6 * N + n;
  float p[3] = { base[0], base[(size_t)N], base[(size_t)2 * N] };
  float q[3] = { base[(size_t)3 * N], base[(size_t)4 * N], base[(size_t)5 * N] };
  xyzt[(size_t)i * 3 + 0] = p[0]; xyzt[(size_t)i * 3 + 1] = p[1]; xyzt[(size_t)i * 3 + 2] = p[2];
  f16* rp = xp + (size_t)i * 32;
  f16* rn = xn + (size_t)i * 32;
  #pragma unroll
  for (int c = 0; c < 32; ++c) {
    rp[c] = (c < 3) ? (f16)p[c] : (f16)0.f;
    rn[c] = (c < 3) ? (f16)q[c] : (f16)0.f;
  }
}

// newxyz[b,s,:] = src[b, fps[s], :]
__global__ void gather3_kernel(const float* __restrict__ src, const int* __restrict__ fps,
                               float* __restrict__ dst, int B, int Np, int S) {
  int i = blockIdx.x * blockDim.x + threadIdx.x;
  if (i >= B * S) return;
  int b = i / S, s = i - b * S;
  int j = fps[s];
  const float* sp = src + (size_t)(b * Np + j) * 3;
  dst[(size_t)i * 3 + 0] = sp[0];
  dst[(size_t)i * 3 + 1] = sp[1];
  dst[(size_t)i * 3 + 2] = sp[2];
}

// KNN (K=16): one block per query, distances in LDS, 16 argmin sweeps.
__global__ __launch_bounds__(128)
void knn_kernel(const float* __restrict__ q, const float* __restrict__ pts,
                unsigned short* __restrict__ idx, int Np, int S) {
  __shared__ float d[1024];
  __shared__ float red[128];
  __shared__ int   redi[128];
  const int qid = blockIdx.x;            // b*S + s
  const int b = qid / S;
  const int tid = threadIdx.x;
  float qx = q[(size_t)qid * 3], qy = q[(size_t)qid * 3 + 1], qz = q[(size_t)qid * 3 + 2];
  for (int n = tid; n < Np; n += blockDim.x) {
    const float* p = pts + (size_t)(b * Np + n) * 3;
    float dx = p[0] - qx, dy = p[1] - qy, dz = p[2] - qz;
    d[n] = dx * dx + dy * dy + dz * dz;
  }
  __syncthreads();
  for (int k = 0; k < 16; ++k) {
    float best = 3.4e38f; int bi = 0;
    for (int n = tid; n < Np; n += blockDim.x)
      if (d[n] < best) { best = d[n]; bi = n; }
    red[tid] = best; redi[tid] = bi;
    __syncthreads();
    for (int st = 64; st > 0; st >>= 1) {
      if (tid < st && red[tid + st] < red[tid]) { red[tid] = red[tid + st]; redi[tid] = redi[tid + st]; }
      __syncthreads();
    }
    if (tid == 0) { idx[(size_t)qid * 16 + k] = (unsigned short)redi[0]; d[redi[0]] = 3.4e38f; }
    __syncthreads();
  }
}

// ---------------------------------------------------------------------------
// Fused SA kernel: one workgroup (8 waves) per query point.
//  TDM gather-mode DMA pulls the 16 KNN rows of pts/cp straight into LDS
//  (pad_interval/pad_amount land rows at stride C+32 f16), ALU fills the 6
//  geometry channels + zero pad, then two WMMA MLP layers per branch with the
//  max-over-16-neighbors done via half-wave shuffles (D-tile columns are the
//  neighbors).
// ---------------------------------------------------------------------------
__global__ __launch_bounds__(256)
void sa_mlp_kernel(const f16* __restrict__ pts, const f16* __restrict__ cp,
                   const float* __restrict__ srcxyz, const float* __restrict__ newxyz,
                   const unsigned short* __restrict__ knn,
                   const f16* __restrict__ w1f, const float* __restrict__ g1f, const float* __restrict__ b1f,
                   const f16* __restrict__ w2f, const float* __restrict__ g2f, const float* __restrict__ b2f,
                   const f16* __restrict__ w1p, const float* __restrict__ g1p, const float* __restrict__ b1p,
                   const f16* __restrict__ w2p, const float* __restrict__ g2p, const float* __restrict__ b2p,
                   f16* __restrict__ featOut, f16* __restrict__ posOut,
                   int C, int c0, int c1, int Np, int S) {
  const int q = blockIdx.x;          // b*S + s
  const int b = q / S;
  const int tid = threadIdx.x, lane = tid & 31, wave = tid >> 5;
  const int nWaves = blockDim.x >> 5;
  const int Kp1 = C + 32;            // Cin = C+6 padded to a multiple of 32

  __shared__ f16 Xs[2 * 16 * 544];   // fits sa4 (C=512 -> Kp1=544), 34 KB
  f16* Xf = Xs;
  f16* Xp = Xs + 16 * Kp1;
  const unsigned short* qi = knn + (size_t)q * 16;

#if USE_TDM
  if (wave == 0) {
    const unsigned int* ip = (const unsigned int*)qi;    // 8 dwords = 16 u16 indices
    i32x4 g2 = { (int)ip[0], (int)ip[1], (int)ip[2], (int)ip[3] };
    i32x4 g3 = { (int)ip[4], (int)ip[5], (int)ip[6], (int)ip[7] };
    // pad after every C f16 (= C/2 dwords) by 16 dwords (=32 f16) -> row stride C+32
    int padInt = (C == 64) ? 4 : (C == 128) ? 5 : (C == 256) ? 6 : 7;
    int d0 = (1 << 16) | (1 << 20) | (padInt << 22) | (15 << 25); // data_size=2B, pad_en
    int d1 = (C & 0xffff) << 16;                                  // tensor_dim0 lo
    int d2 = ((unsigned)C >> 16) | ((Np & 0xffff) << 16);         // dim0 hi | dim1 lo
    int d3 = ((unsigned)Np >> 16) | (C << 16);                    // dim1 hi | tile_dim0
    i32x8 g1v = { d0, d1, d2, d3, 16 /*tile_dim1=#indices*/, C /*dim0 stride*/, 0, 0 };
    unsigned long long gaF = (unsigned long long)(uintptr_t)(pts + (size_t)b * Np * C);
    u32x4 g0F = { 1u | (1u << 31), (unsigned)(uintptr_t)(void*)Xf,
                  (unsigned)gaF, (unsigned)(gaF >> 32) | (2u << 30) };
    tdm_load(g0F, g1v, g2, g3);
    unsigned long long gaP = (unsigned long long)(uintptr_t)(cp + (size_t)b * Np * C);
    u32x4 g0P = { 1u | (1u << 31), (unsigned)(uintptr_t)(void*)Xp,
                  (unsigned)gaP, (unsigned)(gaP >> 32) | (2u << 30) };
    tdm_load(g0P, g1v, g2, g3);
    __builtin_amdgcn_s_wait_tensorcnt(0);
  }
#else
  for (int e = tid; e < 16 * C; e += blockDim.x) {
    int nb = e / C, c = e - nb * C;
    size_t src = (size_t)(b * Np + (int)qi[nb]) * C + c;
    Xf[nb * Kp1 + c] = pts[src];
    Xp[nb * Kp1 + c] = cp[src];
  }
#endif

  // geometry channels C..C+5 = [gxn(3), ctr(3)], C+6..C+31 zero
  float nx = newxyz[(size_t)q * 3], ny = newxyz[(size_t)q * 3 + 1], nz = newxyz[(size_t)q * 3 + 2];
  for (int e = tid; e < 16 * 32; e += blockDim.x) {
    int nb = e >> 5, c = e & 31;
    f16 val = (f16)0.f;
    if (c < 3) {
      const float* p = srcxyz + (size_t)(b * Np + (int)qi[nb]) * 3;
      float ctr = (c == 0) ? nx : (c == 1) ? ny : nz;
      val = (f16)(p[c] - ctr);
    } else if (c < 6) {
      val = (f16)((c == 3) ? nx : (c == 4) ? ny : nz);
    }
    Xf[nb * Kp1 + C + c] = val;
    Xp[nb * Kp1 + C + c] = val;
  }
  __syncthreads();

  for (int br = 0; br < 2; ++br) {
    f16* Xb = br ? Xp : Xf;
    const f16*  W1 = br ? w1p : w1f;  const float* G1 = br ? g1p : g1f;  const float* B1 = br ? b1p : b1f;
    const f16*  W2 = br ? w2p : w2f;  const float* G2 = br ? g2p : g2f;  const float* B2 = br ? b2p : b2f;
    f16* Out = br ? posOut : featOut;

    // layer 1: (c0 x Kp1) * (Kp1 x 16)
    const int nM1 = c0 >> 4, nK1 = Kp1 >> 5;
    v8f acc[4];
    int cnt = 0;
    for (int mt = wave; mt < nM1; mt += nWaves) {
      v8f a = {};
      for (int kt = 0; kt < nK1; ++kt) {
        v16h af = load_a_frag(W1, Kp1, mt << 4, kt << 5, lane);
        v16h bf = load_b_frag(Xb, Kp1, 0, kt << 5, lane);
        a = wmma16(af, bf, a);
      }
      acc[cnt++] = a;
    }
    __syncthreads();                       // all waves done reading Xb
    // bn+lrelu, store H point-major (16 x c0) into the same LDS region
    f16* H = Xb;
    cnt = 0;
    for (int mt = wave; mt < nM1; mt += nWaves) {
      v8f a = acc[cnt++];
      int mOff = (mt << 4) + ((lane & 16) ? 8 : 0);
      v8h hv;
      #pragma unroll
      for (int j = 0; j < 8; ++j) {
        float v = a[j] * (G1[mOff + j] * BN_SCALE_F) + B1[mOff + j];
        hv[j] = (f16)(v >= 0.f ? v : LRELU_SLOPE * v);
      }
      *(v8h*)&H[(lane & 15) * c0 + mOff] = hv;
    }
    __syncthreads();

    // layer 2: (c1 x c0) * (c0 x 16), bn+lrelu, then max over the 16 columns
    const int nM2 = c1 >> 4, nK2 = c0 >> 5;
    for (int mt = wave; mt < nM2; mt += nWaves) {
      v8f a = {};
      for (int kt = 0; kt < nK2; ++kt) {
        v16h af = load_a_frag(W2, c0, mt << 4, kt << 5, lane);
        v16h bf = load_b_frag(H, c0, 0, kt << 5, lane);
        a = wmma16(af, bf, a);
      }
      int mOff = (mt << 4) + ((lane & 16) ? 8 : 0);
      v8h ov;
      #pragma unroll
      for (int j = 0; j < 8; ++j) {
        float v = a[j] * (G2[mOff + j] * BN_SCALE_F) + B2[mOff + j];
        v = v >= 0.f ? v : LRELU_SLOPE * v;
        #pragma unroll
        for (int msk = 1; msk < 16; msk <<= 1)       // max over half-wave = 16 neighbors
          v = fmaxf(v, __shfl_xor(v, msk, 32));
        ov[j] = (f16)v;
      }
      if ((lane & 15) == 0)                          // lane 0 -> m..m+7, lane 16 -> m+8..m+15
        *(v8h*)&Out[(size_t)q * c1 + mOff] = ov;
    }
    __syncthreads();
  }
}

// ---------------------------------------------------------------------------
// CSA epilogue: per point r, channel-softmax attention.
//   qs = sum_c q[c]; e = qs*k; e /= (1e-9 + sum e); a = softmax_c(e)
//   out = feat + a * v
// ---------------------------------------------------------------------------
__global__ __launch_bounds__(256)
void csa_post_kernel(const f16* __restrict__ feat, const f16* __restrict__ q,
                     const f16* __restrict__ k, const f16* __restrict__ v,
                     f16* __restrict__ out, int C) {
  __shared__ float red[256];
  __shared__ float bc[4];
  const int r = blockIdx.x, t = threadIdx.x;
  const f16* qr = q + (size_t)r * C;
  const f16* kr = k + (size_t)r * C;
  const f16* vr = v + (size_t)r * C;
  const f16* fr = feat + (size_t)r * C;

  float s = 0.f;
  for (int c = t; c < C; c += blockDim.x) s += (float)qr[c];
  red[t] = s; __syncthreads();
  for (int st = 128; st > 0; st >>= 1) { if (t < st) red[t] += red[t + st]; __syncthreads(); }
  if (t == 0) bc[0] = red[0];
  __syncthreads();
  const float qs = bc[0];

  s = 0.f;
  for (int c = t; c < C; c += blockDim.x) s += qs * (float)kr[c];
  red[t] = s; __syncthreads();
  for (int st = 128; st > 0; st >>= 1) { if (t < st) red[t] += red[t + st]; __syncthreads(); }
  if (t == 0) bc[1] = 1.f / (1e-9f + red[0]);
  __syncthreads();
  const float inv = bc[1];

  float mx = -3.4e38f;
  for (int c = t; c < C; c += blockDim.x) mx = fmaxf(mx, qs * (float)kr[c] * inv);
  red[t] = mx; __syncthreads();
  for (int st = 128; st > 0; st >>= 1) { if (t < st) red[t] = fmaxf(red[t], red[t + st]); __syncthreads(); }
  if (t == 0) bc[2] = red[0];
  __syncthreads();
  const float m = bc[2];

  s = 0.f;
  for (int c = t; c < C; c += blockDim.x) s += __expf(qs * (float)kr[c] * inv - m);
  red[t] = s; __syncthreads();
  for (int st = 128; st > 0; st >>= 1) { if (t < st) red[t] += red[t + st]; __syncthreads(); }
  if (t == 0) bc[3] = 1.f / red[0];
  __syncthreads();
  const float dinv = bc[3];

  for (int c = t; c < C; c += blockDim.x) {
    float attn = __expf(qs * (float)kr[c] * inv - m) * dinv;
    out[(size_t)r * C + c] = (f16)((float)fr[c] + attn * (float)vr[c]);
  }
}

// y rows (256) = [f(1024) | p(1024)]
__global__ void concat2_kernel(const f16* __restrict__ f, const f16* __restrict__ p,
                               f16* __restrict__ y, int C) {
  int i = blockIdx.x * blockDim.x + threadIdx.x;
  if (i >= 256 * 2 * C) return;
  int r = i / (2 * C), c = i - r * 2 * C;
  y[i] = (c < C) ? f[(size_t)r * C + c] : p[(size_t)r * C + c - C];
}

// (64*4, C) -> (64, C) max over the 4 points
__global__ void maxpool4_kernel(const f16* __restrict__ x, f16* __restrict__ y, int C) {
  int i = blockIdx.x * blockDim.x + threadIdx.x;
  if (i >= 64 * C) return;
  int b = i / C, c = i - b * C;
  float m = -3.4e38f;
  #pragma unroll
  for (int p = 0; p < 4; ++p) m = fmaxf(m, (float)x[(size_t)(b * 4 + p) * C + c]);
  y[i] = (f16)m;
}

// x (64, 48) f16 (rows 40..47 are zero-weight logits) -> out (64, 40) f32 log_softmax
__global__ void logsoftmax_kernel(const f16* __restrict__ x, float* __restrict__ out) {
  __shared__ float sx[40];
  const int r = blockIdx.x, t = threadIdx.x;
  if (t < 40) sx[t] = (float)x[(size_t)r * 48 + t];
  __syncthreads();
  if (t == 0) {
    float m = -3.4e38f;
    for (int c = 0; c < 40; ++c) m = fmaxf(m, sx[c]);
    float s = 0.f;
    for (int c = 0; c < 40; ++c) s += __expf(sx[c] - m);
    float lg = __logf(s);
    for (int c = 0; c < 40; ++c) out[(size_t)r * 40 + c] = sx[c] - m - lg;
  }
}

// ---------------------------------------------------------------------------
// Host orchestration
// Input flattening (setup_inputs insertion order, recursive):
//  0 xyz | 1..6 first_f(w1,g1,b1,w2,g2,b2) | 7..12 first_p
//  sa_l (base=13+18*l): feat mlp +0..5, pos mlp +6..11, csa_f q/k/v +12..14,
//  csa_p q/k/v +15..17 | 85 cat.w 86 cat.g 87 cat.b | 88 fc2 89 g2 90 b2
//  91 fc3 92 g3 93 b3 | 94 fc4 | 95..98 fps s1..s4
// ---------------------------------------------------------------------------
extern "C" void kernel_launch(void* const* d_in, const int* in_sizes, int n_in,
                              void* d_out, int out_size, void* d_ws, size_t ws_size,
                              hipStream_t stream) {
  (void)in_sizes; (void)n_in; (void)out_size; (void)ws_size;
  const int B = 64, N = 1024;
  const float* xyz = (const float*)d_in[0];
  auto F = [&](int i) { return (const float*)d_in[i]; };

  size_t off = 0;
  auto alloc = [&](size_t bytes) -> void* {
    void* p = (char*)d_ws + off;
    off += (bytes + 255) & ~(size_t)255;
    return p;
  };
  auto aH = [&](size_t n) { return (f16*)alloc(n * sizeof(f16)); };
  auto aF = [&](size_t n) { return (float*)alloc(n * sizeof(float)); };
  auto aU = [&](size_t n) { return (unsigned short*)alloc(n * 2); };

  auto cvt = [&](int i, f16* dst, int M, int K, int Mp, int Kp) {
    int n = Mp * Kp;
    cvt_pad_kernel<<<(n + 255) / 256, 256, 0, stream>>>(F(i), dst, M, K, Mp, Kp);
  };
  auto gemm = [&](const f16* W, const f16* X, f16* Y, const float* g, const float* bb,
                  int M, int K, int Ncols, int act) {
    int tiles = (Ncols >> 4) * (M >> 4);
    int blocks = (tiles + 7) >> 3;
    if (blocks > 4096) blocks = 4096;
    if (blocks < 1) blocks = 1;
    wmma_gemm_kernel<<<blocks, 256, 0, stream>>>(W, X, Y, g, bb, M, K, Ncols, act);
  };

  // ---- fp16 (padded) weights ----
  f16* w1f_h = aH(32 * 32);  f16* w2f_h = aH(64 * 32);
  f16* w1p_h = aH(32 * 32);  f16* w2p_h = aH(64 * 32);
  const int Cs[4]  = { 64, 128, 256, 512 };
  const int c0s[4] = { 64, 128, 256, 512 };
  const int c1s[4] = { 128, 256, 512, 1024 };
  const int Ss[4]  = { 256, 64, 16, 4 };
  f16 *sw1f[4], *sw2f[4], *sw1p[4], *sw2p[4], *sqf[4], *skf[4], *svf[4], *sqp[4], *skp[4], *svp[4];
  for (int l = 0; l < 4; ++l) {
    int C = Cs[l], c0 = c0s[l], c1 = c1s[l];
    sw1f[l] = aH((size_t)c0 * (C + 32)); sw2f[l] = aH((size_t)c1 * c0);
    sw1p[l] = aH((size_t)c0 * (C + 32)); sw2p[l] = aH((size_t)c1 * c0);
    sqf[l] = aH((size_t)c1 * c1); skf[l] = aH((size_t)c1 * c1); svf[l] = aH((size_t)c1 * c1);
    sqp[l] = aH((size_t)c1 * c1); skp[l] = aH((size_t)c1 * c1); svp[l] = aH((size_t)c1 * c1);
  }
  f16* cat_h = aH((size_t)1024 * 2048);
  f16* fc2_h = aH((size_t)512 * 1024);
  f16* fc3_h = aH((size_t)256 * 512);
  f16* fc4_h = aH((size_t)48 * 256);           // M padded 40 -> 48

  // ---- activations ----
  f16* xp = aH((size_t)B * N * 32);
  f16* xn = aH((size_t)B * N * 32);
  f16* h1 = aH((size_t)B * N * 32);
  f16* up_p = aH((size_t)B * N * 64);
  f16* up_f = aH((size_t)B * N * 64);
  float* xyzt = aF((size_t)B * N * 3);
  float* nxyz[4]; unsigned short* idxb[4]; f16 *fout[4], *pout[4];
  for (int l = 0; l < 4; ++l) {
    nxyz[l] = aF((size_t)B * Ss[l] * 3);
    idxb[l] = aU((size_t)B * Ss[l] * 16);
    fout[l] = aH((size_t)B * Ss[l] * c1s[l]);
    pout[l] = aH((size_t)B * Ss[l] * c1s[l]);
  }
  const size_t maxPC = (size_t)B * 256 * 128;  // sa1 is the largest (2M elems)
  f16* featpre = aH(maxPC); f16* pospre = aH(maxPC);
  f16* qb = aH(maxPC); f16* kb = aH(maxPC); f16* vb = aH(maxPC);
  f16* xcat = aH((size_t)256 * 2048);
  f16* xc = aH((size_t)256 * 1024);
  f16* xm = aH((size_t)64 * 1024);
  f16* y2 = aH((size_t)64 * 512);
  f16* y3 = aH((size_t)64 * 256);
  f16* y4 = aH((size_t)64 * 48);

  // ---- weight conversions ----
  cvt(1, w1f_h, 32, 3, 32, 32);  cvt(4, w2f_h, 64, 32, 64, 32);
  cvt(7, w1p_h, 32, 3, 32, 32);  cvt(10, w2p_h, 64, 32, 64, 32);
  for (int l = 0; l < 4; ++l) {
    int base = 13 + 18 * l, C = Cs[l], c0 = c0s[l], c1 = c1s[l];
    cvt(base + 0,  sw1f[l], c0, C + 6, c0, C + 32);
    cvt(base + 3,  sw2f[l], c1, c0, c1, c0);
    cvt(base + 6,  sw1p[l], c0, C + 6, c0, C + 32);
    cvt(base + 9,  sw2p[l], c1, c0, c1, c0);
    cvt(base + 12, sqf[l], c1, c1, c1, c1);
    cvt(base + 13, skf[l], c1, c1, c1, c1);
    cvt(base + 14, svf[l], c1, c1, c1, c1);
    cvt(base + 15, sqp[l], c1, c1, c1, c1);
    cvt(base + 16, skp[l], c1, c1, c1, c1);
    cvt(base + 17, svp[l], c1, c1, c1, c1);
  }
  cvt(85, cat_h, 1024, 2048, 1024, 2048);
  cvt(88, fc2_h, 512, 1024, 512, 1024);
  cvt(91, fc3_h, 256, 512, 256, 512);
  cvt(94, fc4_h, 40, 256, 48, 256);

  // ---- stem: transpose + first shared MLPs (norm -> up_f, pos -> up_p) ----
  prep_inputs_kernel<<<(B * N + 255) / 256, 256, 0, stream>>>(xyz, xp, xn, xyzt, B, N);
  gemm(w1p_h, xp, h1, F(8), F(9), 32, 32, B * N, 1);
  gemm(w2p_h, h1, up_p, F(11), F(12), 64, 32, B * N, 1);
  gemm(w1f_h, xn, h1, F(2), F(3), 32, 32, B * N, 1);
  gemm(w2f_h, h1, up_f, F(5), F(6), 64, 32, B * N, 1);

  // ---- 4 set-abstraction layers ----
  const float* srcxyz = xyzt;
  const f16* pts = up_f;
  const f16* cpn = up_p;
  int prevNp = N;
  for (int l = 0; l < 4; ++l) {
    int S = Ss[l], C = Cs[l], c0 = c0s[l], c1 = c1s[l];
    int base = 13 + 18 * l;
    const int* fps = (const int*)d_in[95 + l];
    int Q = B * S;

    gather3_kernel<<<(Q + 255) / 256, 256, 0, stream>>>(srcxyz, fps, nxyz[l], B, prevNp, S);
    knn_kernel<<<Q, 128, 0, stream>>>(nxyz[l], srcxyz, idxb[l], prevNp, S);
    sa_mlp_kernel<<<Q, 256, 0, stream>>>(
        pts, cpn, srcxyz, nxyz[l], idxb[l],
        sw1f[l], F(base + 1), F(base + 2), sw2f[l], F(base + 4), F(base + 5),
        sw1p[l], F(base + 7), F(base + 8), sw2p[l], F(base + 10), F(base + 11),
        featpre, pospre, C, c0, c1, prevNp, S);

    // csa_f: q,k from position (pospre); v from feature (featpre)
    gemm(sqf[l], pospre, qb, nullptr, nullptr, c1, c1, Q, 0);
    gemm(skf[l], pospre, kb, nullptr, nullptr, c1, c1, Q, 0);
    gemm(svf[l], featpre, vb, nullptr, nullptr, c1, c1, Q, 0);
    csa_post_kernel<<<Q, 256, 0, stream>>>(featpre, qb, kb, vb, fout[l], c1);
    // csa_p: feature = pospre, position = featpre
    gemm(sqp[l], featpre, qb, nullptr, nullptr, c1, c1, Q, 0);
    gemm(skp[l], featpre, kb, nullptr, nullptr, c1, c1, Q, 0);
    gemm(svp[l], pospre, vb, nullptr, nullptr, c1, c1, Q, 0);
    csa_post_kernel<<<Q, 256, 0, stream>>>(pospre, qb, kb, vb, pout[l], c1);

    srcxyz = nxyz[l]; pts = fout[l]; cpn = pout[l]; prevNp = S;
  }

  // ---- head ----
  concat2_kernel<<<(256 * 2048 + 255) / 256, 256, 0, stream>>>(fout[3], pout[3], xcat, 1024);
  gemm(cat_h, xcat, xc, F(86), F(87), 1024, 2048, 256, 1);
  maxpool4_kernel<<<(64 * 1024 + 255) / 256, 256, 0, stream>>>(xc, xm, 1024);
  gemm(fc2_h, xm, y2, F(89), F(90), 512, 1024, 64, 1);
  gemm(fc3_h, y2, y3, F(92), F(93), 256, 512, 64, 1);
  gemm(fc4_h, y3, y4, nullptr, nullptr, 48, 256, 64, 0);
  logsoftmax_kernel<<<64, 64, 0, stream>>>(y4, (float*)d_out);
}
